// Encoder_23905787969975
// MI455X (gfx1250) — matrix operand order repaired and checked by reading, not compiled, not run on hardware
//
#include <hip/hip_runtime.h>
#include <hip/hip_bf16.h>

typedef __attribute__((ext_vector_type(16))) __bf16 v16bf;
typedef __attribute__((ext_vector_type(8)))  float  v8f;

#define DM   1024   // d_model
#define NH   16     // heads
#define DKH  64     // d_k per head
#define BQ   256    // query rows per block (8 waves x 2 tiles x 16)

__global__ __launch_bounds__(256)
void attn_layer_kernel(const float* __restrict__ xin,
                       const float* __restrict__ w6,   // this layer: [NH][6]
                       const int* __restrict__ pN,
                       const int* __restrict__ pK,
                       float* __restrict__ xout,
                       int B, int L)
{
  __shared__ __bf16 lds_kv [32][DKH];   // K/V tile, row-major (j, dk)
  __shared__ __bf16 lds_kvt[DKH][32];   // transposed (dk, j)
  __shared__ __bf16 lds_p  [8][32][32]; // per-wave P tiles (2x16 rows, j_local)

  const int  tid  = threadIdx.x;
  const int  wave = tid >> 5;
  const int  lane = tid & 31;
  const int  lh   = lane & 15;
  const bool hiHalf = lane >= 16;
  const int  hoff = hiHalf ? 8 : 0;

  const int h  = blockIdx.y;
  const int b  = blockIdx.z;
  const int i0 = blockIdx.x * BQ + wave * 32;   // this wave: rows [i0, i0+32)

  const int Kc = pK[0];
  const int NK = pN[0] * Kc;

  // per-head tanh(class weights)
  float tw[6];
#pragma unroll
  for (int t = 0; t < 6; ++t) tw[t] = tanhf(w6[h * 6 + t]);

  // per-row constants hoisted out of the J loop (row class-block id)
  int idv[2][8];
#pragma unroll
  for (int t = 0; t < 2; ++t)
#pragma unroll
    for (int r = 0; r < 8; ++r)
      idv[t][r] = (i0 + t * 16 + r + hoff) / Kc;

  // ---- Q tiles in WMMA A-layout (2 row tiles x two K=32 chunks) ----
  const int abase = hiHalf ? 16 : 0;
  v16bf Aq[2][2];
#pragma unroll
  for (int t = 0; t < 2; ++t) {
    int qrow = i0 + t * 16 + lh;              // A layout: lane -> row M = lane%16
    int qr   = (qrow < L) ? qrow : (L - 1);   // clamp OOB rows (never stored)
    const float* qptr = xin + ((size_t)b * L + qr) * DM + h * DKH;
#pragma unroll
    for (int p = 0; p < 8; ++p) {             // VGPR p holds K pair {k0, k0+1}
      int k0 = 2 * p + abase;
      Aq[t][0][2 * p]     = (__bf16)qptr[k0];
      Aq[t][0][2 * p + 1] = (__bf16)qptr[k0 + 1];
      Aq[t][1][2 * p]     = (__bf16)qptr[32 + k0];
      Aq[t][1][2 * p + 1] = (__bf16)qptr[32 + k0 + 1];
    }
  }

  // flash-attention running state; register r -> row M = r + hoff (per tile)
  float m_r[2][8], l_r[2][8];
  v8f acc[2][4];
  const v8f vzero = {};
#pragma unroll
  for (int t = 0; t < 2; ++t)
#pragma unroll
    for (int r = 0; r < 8; ++r) { m_r[t][r] = -1e30f; l_r[t][r] = 0.0f; }
#pragma unroll
  for (int t = 0; t < 2; ++t)
#pragma unroll
    for (int n = 0; n < 4; ++n) acc[t][n] = vzero;

  const float scale = 0.125f;  // 1/sqrt(64)

  for (int j0 = 0; j0 < L; j0 += 32) {
    __syncthreads();  // previous iteration's LDS reads complete
    // ---- cooperative stage: 32x64 f32 -> bf16 tile + transposed copy ----
#pragma unroll
    for (int it = 0; it < 2; ++it) {
      int idx = tid + it * 256;     // float4 index, 512 total
      int jl  = idx >> 4;           // 0..31
      int c4  = (idx & 15) << 2;    // 0..60
      int jg  = j0 + jl;
      int jcl = (jg < L) ? jg : (L - 1);   // clamp: OOB cols are score-masked
      const float* src = xin + ((size_t)b * L + jcl) * DM + h * DKH + c4;
      float4 v = *(const float4*)src;
      // prefetch next tile's rows into L2 while this tile computes
      int jpre = jg + 32; jpre = (jpre < L) ? jpre : (L - 1);
      __builtin_prefetch(xin + ((size_t)b * L + jpre) * DM + h * DKH + c4, 0, 1);
      __bf16 e0 = (__bf16)v.x, e1 = (__bf16)v.y, e2 = (__bf16)v.z, e3 = (__bf16)v.w;
      lds_kv[jl][c4 + 0] = e0; lds_kv[jl][c4 + 1] = e1;
      lds_kv[jl][c4 + 2] = e2; lds_kv[jl][c4 + 3] = e3;
      lds_kvt[c4 + 0][jl] = e0; lds_kvt[c4 + 1][jl] = e1;
      lds_kvt[c4 + 2][jl] = e2; lds_kvt[c4 + 3][jl] = e3;
    }
    __syncthreads();

    // ---- per-J-step lane constants (hoisted out of row loops) ----
    const int  jc0 = j0 + lh, jc1 = jc0 + 16;
    const bool ok0 = jc0 < L, ok1 = jc1 < L;
    const bool jge0 = jc0 >= NK, jge1 = jc1 >= NK;
    const int  jd0 = jc0 / Kc, jd1 = jc1 / Kc;

    // ---- B operands for S = Q K^T (shared by both row tiles) ----
    const int bb = hiHalf ? 16 : 0;
    v16bf B00, B01, B10, B11;
#pragma unroll
    for (int t = 0; t < 16; ++t) {
      B00[t] = lds_kv[lh][bb + t];            // cols j0..j0+15, dk 0..31
      B01[t] = lds_kv[lh][32 + bb + t];       // cols j0..j0+15, dk 32..63
      B10[t] = lds_kv[16 + lh][bb + t];       // cols j0+16..j0+31
      B11[t] = lds_kv[16 + lh][32 + bb + t];
    }

#pragma unroll
    for (int t = 0; t < 2; ++t) {
      v8f S0 = vzero, S1 = vzero;
      S0 = __builtin_amdgcn_wmma_f32_16x16x32_bf16(false, Aq[t][0], false, B00, (short)0, S0, false, false);
      S0 = __builtin_amdgcn_wmma_f32_16x16x32_bf16(false, Aq[t][1], false, B01, (short)0, S0, false, false);
      S1 = __builtin_amdgcn_wmma_f32_16x16x32_bf16(false, Aq[t][0], false, B10, (short)0, S1, false, false);
      S1 = __builtin_amdgcn_wmma_f32_16x16x32_bf16(false, Aq[t][1], false, B11, (short)0, S1, false, false);

      // ---- online softmax + structured weight; write P tile (bf16) ----
#pragma unroll
      for (int r = 0; r < 8; ++r) {
        float s0 = ok0 ? S0[r] * scale : -1e30f;
        float s1 = ok1 ? S1[r] * scale : -1e30f;
        float mx = fmaxf(s0, s1);
#pragma unroll
        for (int msk = 1; msk < 16; msk <<= 1)        // stays within 16-lane half
          mx = fmaxf(mx, __shfl_xor(mx, msk, 32));
        float nm = fmaxf(m_r[t][r], mx);
        float cf = __expf(m_r[t][r] - nm);
        m_r[t][r] = nm;
        l_r[t][r] *= cf;
#pragma unroll
        for (int n = 0; n < 4; ++n) acc[t][n][r] *= cf;

        float e0 = __expf(s0 - nm);
        float e1 = __expf(s1 - nm);
        float sum = e0 + e1;
#pragma unroll
        for (int msk = 1; msk < 16; msk <<= 1)
          sum += __shfl_xor(sum, msk, 32);
        l_r[t][r] += sum;                              // unweighted denominator

        // structured class weight: cheap selects only (divisions hoisted)
        int  ir  = i0 + t * 16 + r + hoff;
        bool ig  = ir >= NK;
        float w0 = ig ? (jge0 ? tw[5] : tw[4])
                      : (jge0 ? tw[3] : ((ir == jc0) ? tw[0] : ((idv[t][r] == jd0) ? tw[1] : tw[2])));
        float w1 = ig ? (jge1 ? tw[5] : tw[4])
                      : (jge1 ? tw[3] : ((ir == jc1) ? tw[0] : ((idv[t][r] == jd1) ? tw[1] : tw[2])));

        int M = t * 16 + r + hoff;
        lds_p[wave][M][lh]      = (__bf16)(e0 * w0);
        lds_p[wave][M][16 + lh] = (__bf16)(e1 * w1);
      }
    }
    __syncthreads();  // P visible (and orders vs. next staging)

    // ---- O += P @ V : P back in A-layout; V B-operand reused across tiles ----
    v16bf PA[2];
#pragma unroll
    for (int t = 0; t < 2; ++t)
#pragma unroll
      for (int p = 0; p < 8; ++p) {
        int k0 = 2 * p + abase;
        PA[t][2 * p]     = lds_p[wave][t * 16 + lh][k0];
        PA[t][2 * p + 1] = lds_p[wave][t * 16 + lh][k0 + 1];
      }
#pragma unroll
    for (int n = 0; n < 4; ++n) {
      v16bf BV;
#pragma unroll
      for (int u = 0; u < 16; ++u)
        BV[u] = lds_kvt[n * 16 + lh][bb + u];          // n = dk tile, contiguous j
      acc[0][n] = __builtin_amdgcn_wmma_f32_16x16x32_bf16(false, PA[0], false, BV, (short)0, acc[0][n], false, false);
      acc[1][n] = __builtin_amdgcn_wmma_f32_16x16x32_bf16(false, PA[1], false, BV, (short)0, acc[1][n], false, false);
    }
  }

  // ---- normalize and store (head h slice of xout) ----
#pragma unroll
  for (int t = 0; t < 2; ++t)
#pragma unroll
    for (int n = 0; n < 4; ++n)
#pragma unroll
      for (int r = 0; r < 8; ++r) {
        int i = i0 + t * 16 + r + hoff;
        if (i < L)
          xout[((size_t)b * L + i) * DM + h * DKH + n * 16 + lh] = acc[t][n][r] / l_r[t][r];
      }
}

extern "C" void kernel_launch(void* const* d_in, const int* in_sizes, int n_in,
                              void* d_out, int out_size, void* d_ws, size_t ws_size,
                              hipStream_t stream) {
  const float* samples = (const float*)d_in[0];   // [B, L, DM] f32
  const float* attw    = (const float*)d_in[1];   // [2, NH, 6] f32
  const int*   pN      = (const int*)d_in[2];
  const int*   pK      = (const int*)d_in[3];
  float*       out     = (float*)d_out;
  float*       xmid    = (float*)d_ws;            // layer-0 output, B*L*DM f32

  const int B = 8;
  const int L = in_sizes[0] / (B * DM);           // 1001

  dim3 grid((L + BQ - 1) / BQ, NH, B);
  dim3 blk(256);
  // layer 0: samples -> xmid
  attn_layer_kernel<<<grid, blk, 0, stream>>>(samples, attw + 0 * NH * 6, pN, pK, xmid, B, L);
  // layer 1: xmid -> out
  attn_layer_kernel<<<grid, blk, 0, stream>>>(xmid, attw + 1 * NH * 6, pN, pK, out, B, L);
}
